// TransConvLayer_8495445311658
// MI455X (gfx1250) — compile-verified, matrix operand order checked by reading, not run on hardware
//
#include <hip/hip_runtime.h>

typedef __attribute__((ext_vector_type(16))) __bf16 v16bf;
typedef __attribute__((ext_vector_type(8)))  float  v8f;

#define N_NODES 100000
#define C_IN    256
#define H_HEADS 4
#define D_HEAD  128
#define HD      512
#define RB      32            // rows per block tile (100000 = 3125 * 32, exact)
#define NB      3125
#define NBX     125           // persistent WGs along x in pass 1

// workspace float-region layout (then packed bf16 weights)
#define ACC_SSQ        0
#define ACC_SSK        1
#define ACC_KSUM       16                  // 512 floats
#define ACC_KVS        1024                // 4*128*128 floats (fragment-packed)
#define ACC_F32_FLOATS 66560
#define WQ16_BYTE_OFF  (ACC_F32_FLOATS * 4)   // 266240 bytes
#define W_ELEMS        (C_IN * HD)            // 131072

// ---------------------------------------------------------------------------
// Fragment-packed layout.
//
// WMMA bf16 operand layout (ISA 7.12.2, wave32): for A (16x32, MxK) lane L
// holds row M=L&15 with half=L>>4 selecting K-subranges; element e of the
// v16bf maps to k(e) = 2*((e>>1)&3) + ((e>>1)&4 ? 16:0) + 8*half + (e&1).
// B (32x16, KxN) mirrors this with N=L&15.
//
// "Packed" storage keeps each lane's 16 elements contiguous:
//    tile_base[lane*16 + e]
// so a fragment load is one 32B/lane contiguous vector load.
//
// Producer side: for the projection's C-layout (k = mt*16 + rbase + r,
// rbase = (lane>>4)*8) the destination is the SAME lane, element e = mt*8+r,
// i.e. each thread writes 8 consecutive elements at tile*512 + lane*16 + mt*8.
// ---------------------------------------------------------------------------
__device__ __forceinline__ v8f zero_v8() {
  v8f z;
#pragma unroll
  for (int i = 0; i < 8; ++i) z[i] = 0.0f;
  return z;
}

// row-major A fragment (used for astage / qtile staging buffers)
__device__ __forceinline__ v16bf load_frag_a(const __bf16* base, int ld) {
  const int lane = threadIdx.x & 31;
  const int m = lane & 15;
  const int h8 = (lane >> 4) * 8;
  v16bf f;
#pragma unroll
  for (int i = 0; i < 8; ++i) {
    const int k = ((i & 3) * 2) + ((i & 4) ? 16 : 0) + h8;
    f[2 * i]     = base[m * ld + k];
    f[2 * i + 1] = base[m * ld + k + 1];
  }
  return f;
}

// fragment-packed bf16 load (global or LDS): 32 contiguous bytes per lane
__device__ __forceinline__ v16bf load_frag_packed(const __bf16* tile) {
  return *(const v16bf*)(tile + (threadIdx.x & 31) * 16);
}

// fragment-packed f32 load + convert (for kvs workspace)
__device__ __forceinline__ v16bf load_frag_packed_f32(const float* tile) {
  const float* src = tile + (threadIdx.x & 31) * 16;
  v8f lo = *(const v8f*)src;
  v8f hi = *(const v8f*)(src + 8);
  v16bf f;
#pragma unroll
  for (int j = 0; j < 8; ++j) {
    f[j]     = (__bf16)lo[j];
    f[8 + j] = (__bf16)hi[j];
  }
  return f;
}

__device__ __forceinline__ v8f wmma_bf16(v16bf a, v16bf b, v8f c) {
  return __builtin_amdgcn_wmma_f32_16x16x32_bf16(false, a, false, b, (short)0, c,
                                                 false, false);
}

// ---------------------------------------------------------------------------
// Convert fp32 weights -> fragment-packed bf16 copies in workspace.
// Output element o = (kb*32 + ct)*512 + lane*16 + e  holds W[k, ct*16+n] with
// n = lane&15, half = lane>>4, k = kb*32 + 2*((e>>1)&3) + ((e>>1)&4?16:0)
//                                 + 8*half + (e&1).
// ---------------------------------------------------------------------------
__global__ __launch_bounds__(256) void k_cvt_weights(
    const float* __restrict__ Wq, const float* __restrict__ Wk,
    const float* __restrict__ Wv, __bf16* __restrict__ wq16,
    __bf16* __restrict__ wk16, __bf16* __restrict__ wv16) {
  const int stride = gridDim.x * blockDim.x;
  for (int o = blockIdx.x * blockDim.x + threadIdx.x; o < W_ELEMS; o += stride) {
    const int e = o & 15;
    const int L = (o >> 4) & 31;
    const int blk = o >> 9;      // 0..255
    const int ct = blk & 31;     // column tile (16 cols)
    const int kb = blk >> 5;     // k block (32 rows)
    const int i = e >> 1;
    const int k = kb * 32 + ((i & 3) * 2) + ((i & 4) ? 16 : 0) + (L >> 4) * 8 +
                  (e & 1);
    const int col = ct * 16 + (L & 15);
    wq16[o] = (__bf16)Wq[k * HD + col];
    wk16[o] = (__bf16)Wk[k * HD + col];
    wv16[o] = (__bf16)Wv[k * HD + col];
  }
}

// ---------------------------------------------------------------------------
// Pass 1: per (x-slice, head) persistent workgroup.
// K,V,Q projections with WMMA; accumulates sum(q^2), sum(k^2), ks_sum and
// kvs = K^T V (per head, register fragments, fragment-packed flush).
// ---------------------------------------------------------------------------
__global__ __launch_bounds__(256) void k_proj_stats(
    const float* __restrict__ qin, const float* __restrict__ kin,
    const float* __restrict__ vin, const float* __restrict__ bq,
    const float* __restrict__ bk, const float* __restrict__ bv,
    const __bf16* __restrict__ wq16, const __bf16* __restrict__ wk16,
    const __bf16* __restrict__ wv16, float* __restrict__ acc) {
  __shared__ __bf16 astage[RB * C_IN];   // 16 KB, reused for K/V/Q inputs
  __shared__ __bf16 ktp[8 * 512];        // 8 KB, fragment-packed K tile (A of K^T V)
  __shared__ __bf16 vtp[8 * 512];        // 8 KB, fragment-packed V tile (B of K^T V)
  __shared__ float redq[8];
  __shared__ float redk[8];

  const int tid = threadIdx.x;
  const int wave = tid >> 5;
  const int lane = tid & 31;
  const int nl = lane & 15;
  const int rbase = (lane >> 4) * 8;
  const int head = blockIdx.y;
  const int col0 = head * D_HEAD;
  const int mt = wave & 1;
  const int ntb = (wave >> 1) * 2;
  const int ct0 = head * 8 + ntb;        // global column tile of this wave's 1st tile

  v8f kv_acc[8];
#pragma unroll
  for (int j = 0; j < 8; ++j) kv_acc[j] = zero_v8();
  float ssq = 0.0f, ssk = 0.0f, kcs0 = 0.0f, kcs1 = 0.0f;

  for (int nb = blockIdx.x; nb < NB; nb += NBX) {
    // Opaque *offset* (always 0): keeps the weight fragments loop-variant so
    // LICM cannot hoist (and spill) them across the nb loop, while the base
    // pointers remain kernarg-derived so loads lower to global_load (not
    // flat_load, which would also tie up DScnt / the LDS return path).
    size_t woff = 0;
    asm volatile("" : "+s"(woff));
    const __bf16* wq = wq16 + woff;
    const __bf16* wk = wk16 + woff;
    const __bf16* wv = wv16 + woff;

    const size_t rowoff = (size_t)nb * RB * C_IN;
    __syncthreads();  // protect astage/ktp/vtp from previous iteration

    // ---- stage K input rows (fp32 -> bf16), prefetch next slice ----
    if (nb + NBX < NB)
      __builtin_prefetch(kin + rowoff + (size_t)NBX * RB * C_IN + tid * 32, 0, 3);
    for (int j = tid; j < RB * C_IN; j += 256) astage[j] = (__bf16)kin[rowoff + j];
    __syncthreads();

    // ---- K projection: [32x256]@[256x128] -> ktp (+bk), ssk, ks col sums ----
    {
      v8f c0 = zero_v8(), c1 = zero_v8();
#pragma unroll 2
      for (int kk = 0; kk < C_IN; kk += 32) {
        v16bf a = load_frag_a(astage + mt * 16 * C_IN + kk, C_IN);
        v16bf b0 = load_frag_packed(wk + ((size_t)(kk >> 5) * 32 + ct0) * 512);
        v16bf b1 = load_frag_packed(wk + ((size_t)(kk >> 5) * 32 + ct0 + 1) * 512);
        c0 = wmma_bf16(a, b0, c0);
        c1 = wmma_bf16(a, b1, c1);
      }
      const float bias0 = bk[col0 + ntb * 16 + nl];
      const float bias1 = bk[col0 + (ntb + 1) * 16 + nl];
      __bf16* d0 = ktp + ntb * 512 + lane * 16 + mt * 8;
      __bf16* d1 = ktp + (ntb + 1) * 512 + lane * 16 + mt * 8;
#pragma unroll
      for (int r = 0; r < 8; ++r) {
        const float x0 = c0[r] + bias0;
        const float x1 = c1[r] + bias1;
        ssk += x0 * x0 + x1 * x1;
        kcs0 += x0;
        kcs1 += x1;
        d0[r] = (__bf16)x0;   // 8 consecutive bf16 -> vector ds_store
        d1[r] = (__bf16)x1;
      }
    }
    __syncthreads();

    // ---- stage V input rows ----
    if (nb + NBX < NB)
      __builtin_prefetch(vin + rowoff + (size_t)NBX * RB * C_IN + tid * 32, 0, 3);
    for (int j = tid; j < RB * C_IN; j += 256) astage[j] = (__bf16)vin[rowoff + j];
    __syncthreads();

    // ---- V projection -> vtp (+bv), fragment-packed ----
    {
      v8f c0 = zero_v8(), c1 = zero_v8();
#pragma unroll 2
      for (int kk = 0; kk < C_IN; kk += 32) {
        v16bf a = load_frag_a(astage + mt * 16 * C_IN + kk, C_IN);
        v16bf b0 = load_frag_packed(wv + ((size_t)(kk >> 5) * 32 + ct0) * 512);
        v16bf b1 = load_frag_packed(wv + ((size_t)(kk >> 5) * 32 + ct0 + 1) * 512);
        c0 = wmma_bf16(a, b0, c0);
        c1 = wmma_bf16(a, b1, c1);
      }
      const float bias0 = bv[col0 + ntb * 16 + nl];
      const float bias1 = bv[col0 + (ntb + 1) * 16 + nl];
      __bf16* d0 = vtp + ntb * 512 + lane * 16 + mt * 8;
      __bf16* d1 = vtp + (ntb + 1) * 512 + lane * 16 + mt * 8;
#pragma unroll
      for (int r = 0; r < 8; ++r) {
        d0[r] = (__bf16)(c0[r] + bias0);
        d1[r] = (__bf16)(c1[r] + bias1);
      }
    }
    __syncthreads();

    // ---- stage Q input rows ----
    if (nb + NBX < NB)
      __builtin_prefetch(qin + rowoff + (size_t)NBX * RB * C_IN + tid * 32, 0, 3);
    for (int j = tid; j < RB * C_IN; j += 256) astage[j] = (__bf16)qin[rowoff + j];
    __syncthreads();

    // ---- Q projection: only need sum(q^2) ----
    {
      v8f c0 = zero_v8(), c1 = zero_v8();
#pragma unroll 2
      for (int kk = 0; kk < C_IN; kk += 32) {
        v16bf a = load_frag_a(astage + mt * 16 * C_IN + kk, C_IN);
        v16bf b0 = load_frag_packed(wq + ((size_t)(kk >> 5) * 32 + ct0) * 512);
        v16bf b1 = load_frag_packed(wq + ((size_t)(kk >> 5) * 32 + ct0 + 1) * 512);
        c0 = wmma_bf16(a, b0, c0);
        c1 = wmma_bf16(a, b1, c1);
      }
      const float bias0 = bq[col0 + ntb * 16 + nl];
      const float bias1 = bq[col0 + (ntb + 1) * 16 + nl];
#pragma unroll
      for (int r = 0; r < 8; ++r) {
        const float x0 = c0[r] + bias0;
        const float x1 = c1[r] + bias1;
        ssq += x0 * x0 + x1 * x1;
      }
    }

    // ---- kvs += K^T V  (M=128, N=128, K=32); wave owns m-tile = wave ----
    {
      v16bf at = load_frag_packed(ktp + wave * 512);
#pragma unroll
      for (int j = 0; j < 8; ++j) {
        v16bf b = load_frag_packed(vtp + j * 512);
        kv_acc[j] = wmma_bf16(at, b, kv_acc[j]);
      }
    }
  }

  // ---- flush scalar reductions ----
#pragma unroll
  for (int off = 16; off > 0; off >>= 1) {
    ssq += __shfl_down(ssq, off, 32);
    ssk += __shfl_down(ssk, off, 32);
  }
  if (lane == 0) { redq[wave] = ssq; redk[wave] = ssk; }
  __syncthreads();
  if (tid == 0) {
    float a = 0.0f, b = 0.0f;
    for (int i = 0; i < 8; ++i) { a += redq[i]; b += redk[i]; }
    atomicAdd(&acc[ACC_SSQ], a);
    atomicAdd(&acc[ACC_SSK], b);
  }
  atomicAdd(&acc[ACC_KSUM + col0 + ntb * 16 + nl], kcs0);
  atomicAdd(&acc[ACC_KSUM + col0 + (ntb + 1) * 16 + nl], kcs1);

  // ---- flush kvs accumulators, fragment-packed f32 layout ----
  // D-layout element (r): m = wave*16 + rbase + r, d = j*16 + nl.
  // Packed index: block = (m>>5)*8 + j = (wave>>1)*8 + j; lane' = lane;
  // elem = (wave&1)*8 + r  -> 8 consecutive floats per (j).
  float* kvg = acc + ACC_KVS + head * D_HEAD * D_HEAD;
#pragma unroll
  for (int j = 0; j < 8; ++j) {
    float* dst = kvg + ((wave >> 1) * 8 + j) * 512 + lane * 16 + (wave & 1) * 8;
#pragma unroll
    for (int r = 0; r < 8; ++r) atomicAdd(&dst[r], kv_acc[j][r]);
  }
}

// ---------------------------------------------------------------------------
// Pass 2: recompute Q (bf16 LDS tile), then per 16-column tile j: recompute
// V fragment, compute attention with WMMA, combine and write out.
// out[row,d] = mean_h (s * (q @ kvs_raw)[h,d] + N*v[h,d]) / (s * q.ks_sum[h] + N)
// with s = 1/(||q|| * ||k||) folded out of both normalizations.
// ---------------------------------------------------------------------------
__global__ __launch_bounds__(256) void k_output(
    const float* __restrict__ qin, const float* __restrict__ vin,
    const float* __restrict__ bq, const float* __restrict__ bv,
    const __bf16* __restrict__ wq16, const __bf16* __restrict__ wv16,
    const float* __restrict__ acc, float* __restrict__ out) {
  __shared__ __bf16 astage[RB * C_IN];  // 16 KB
  __shared__ __bf16 qtile[RB * HD];     // 32 KB (row-major; feeds A frags + dot)
  __shared__ float normd[RB * H_HEADS];

  const int tid = threadIdx.x;
  const int wave = tid >> 5;
  const int lane = tid & 31;
  const int nl = lane & 15;
  const int rbase = (lane >> 4) * 8;
  const int mt = wave & 1;    // row half of the 32-row block
  const int h = wave >> 1;    // head owned by this wave
  const size_t r0 = (size_t)blockIdx.x * RB;
  const size_t rowoff = r0 * C_IN;

  const float s = rsqrtf(acc[ACC_SSQ]) * rsqrtf(acc[ACC_SSK]);
  const float Nf = (float)N_NODES;

  // ---- stage Q input, project to qtile (bf16, biased) ----
  for (int j = tid; j < RB * C_IN; j += 256) astage[j] = (__bf16)qin[rowoff + j];
  __syncthreads();
  {
    v8f qc[8];
#pragma unroll
    for (int j = 0; j < 8; ++j) qc[j] = zero_v8();
#pragma unroll 1
    for (int kk = 0; kk < C_IN; kk += 32) {
      v16bf a = load_frag_a(astage + mt * 16 * C_IN + kk, C_IN);
#pragma unroll
      for (int j = 0; j < 8; ++j) {
        v16bf b =
            load_frag_packed(wq16 + ((size_t)(kk >> 5) * 32 + h * 8 + j) * 512);
        qc[j] = wmma_bf16(a, b, qc[j]);
      }
    }
#pragma unroll
    for (int j = 0; j < 8; ++j) {
      const int col = (h * 8 + j) * 16 + nl;
      const float bb = bq[col];
#pragma unroll
      for (int r = 0; r < 8; ++r)
        qtile[(mt * 16 + rbase + r) * HD + col] = (__bf16)(qc[j][r] + bb);
    }
  }
  __syncthreads();

  // ---- stage V input rows ----
  for (int j = tid; j < RB * C_IN; j += 256) astage[j] = (__bf16)vin[rowoff + j];
  __syncthreads();

  // ---- normalizer: denom[row][h] = s * (q . ks_sum) + N ----
  if (tid < RB * H_HEADS) {
    const int row = tid >> 2;
    const int hh = tid & 3;
    float d = 0.0f;
#pragma unroll 4
    for (int m = 0; m < D_HEAD; ++m)
      d += (float)qtile[row * HD + hh * D_HEAD + m] * acc[ACC_KSUM + hh * D_HEAD + m];
    normd[row * H_HEADS + hh] = s * d + Nf;
  }
  __syncthreads();

  // ---- per column tile j: V projection + attention + combine + writeout ----
  const float* kvg = acc + ACC_KVS + h * D_HEAD * D_HEAD;
#pragma unroll 1
  for (int j = 0; j < 8; ++j) {
    // V projection for this 16-col tile
    v8f vcj = zero_v8();
#pragma unroll 2
    for (int kk = 0; kk < C_IN; kk += 32) {
      v16bf a = load_frag_a(astage + mt * 16 * C_IN + kk, C_IN);
      v16bf b =
          load_frag_packed(wv16 + ((size_t)(kk >> 5) * 32 + h * 8 + j) * 512);
      vcj = wmma_bf16(a, b, vcj);
    }
    const float bb = bv[(h * 8 + j) * 16 + nl];

    // attention: q_h @ kvs_h for this 16-col tile (K = 128)
    v8f attnj = zero_v8();
#pragma unroll 2
    for (int kk = 0; kk < D_HEAD; kk += 32) {
      v16bf a = load_frag_a(qtile + mt * 16 * HD + h * D_HEAD + kk, HD);
      v16bf b = load_frag_packed_f32(kvg + ((kk >> 5) * 8 + j) * 512);
      attnj = wmma_bf16(a, b, attnj);
    }

    // combine and accumulate head-mean into d_out (pre-zeroed)
#pragma unroll
    for (int r = 0; r < 8; ++r) {
      const int row = mt * 16 + rbase + r;
      const float num = s * attnj[r] + Nf * (vcj[r] + bb);
      const float val = 0.25f * num / normd[row * H_HEADS + h];
      atomicAdd(&out[(r0 + row) * D_HEAD + j * 16 + nl], val);
    }
  }
}

// ---------------------------------------------------------------------------
extern "C" void kernel_launch(void* const* d_in, const int* in_sizes, int n_in,
                              void* d_out, int out_size, void* d_ws, size_t ws_size,
                              hipStream_t stream) {
  const float* qin = (const float*)d_in[0];
  const float* kin = (const float*)d_in[1];
  const float* vin = (const float*)d_in[2];
  const float* Wq  = (const float*)d_in[3];
  const float* bq  = (const float*)d_in[4];
  const float* Wk  = (const float*)d_in[5];
  const float* bk  = (const float*)d_in[6];
  const float* Wv  = (const float*)d_in[7];
  const float* bv  = (const float*)d_in[8];
  float* out = (float*)d_out;

  float* acc = (float*)d_ws;
  __bf16* wq16 = (__bf16*)((char*)d_ws + WQ16_BYTE_OFF);
  __bf16* wk16 = wq16 + W_ELEMS;
  __bf16* wv16 = wk16 + W_ELEMS;

  // zero accumulators + output (atomically accumulated into)
  hipMemsetAsync(acc, 0, ACC_F32_FLOATS * sizeof(float), stream);
  hipMemsetAsync(out, 0, (size_t)out_size * sizeof(float), stream);

  k_cvt_weights<<<512, 256, 0, stream>>>(Wq, Wk, Wv, wq16, wk16, wv16);

  dim3 g1(NBX, H_HEADS);
  k_proj_stats<<<g1, 256, 0, stream>>>(qin, kin, vin, bq, bk, bv, wq16, wk16, wv16,
                                       acc);

  k_output<<<NB, 256, 0, stream>>>(qin, vin, bq, bv, wq16, wv16, acc, out);
}
